// ViT5DeformableTransformerDecoderLayer_60335700574223
// MI455X (gfx1250) — compile-verified, matrix operand order checked by reading, not run on hardware
//
#include <hip/hip_runtime.h>
#include <hip/hip_bf16.h>
#include <math.h>

// ---------------------------------------------------------------------------
// Problem constants (fixed by the reference module)
// ---------------------------------------------------------------------------
#define D     256
#define NH    8
#define HD    32
#define LL    4
#define PP    4
#define NQ    900
#define BS    8
#define DFFN  1024
#define NVPX  21760            // total value pixels over 4 levels
#define NTOK  (NQ * BS)        // 7200 decoder tokens
#define NVTOK (NVPX * BS)      // 174080 memory tokens
#define QBLK  57               // ceil(900/16)

typedef __attribute__((ext_vector_type(16))) _Float16 v16h;
typedef __attribute__((ext_vector_type(8)))  float    v8f;
typedef __attribute__((ext_vector_type(4)))  unsigned int u32x4;
typedef __attribute__((ext_vector_type(8)))  int      i32x8;
typedef __attribute__((ext_vector_type(4)))  int      i32x4;

// ---------------------------------------------------------------------------
// WMMA fragment loader (CDNA5 ISA 7.12.2 layout).
//   lanes 0-15 : row = lane,    K chunks {k0+0..7,  k0+16..23}
//   lanes 16-31: row = lane-16, K chunks {k0+8..15, k0+24..31}
// B (KxN) column n == row n of W since we compute X @ W^T -> same loader.
// Works for global f16/f32 and for generic (LDS) pointers.
// ---------------------------------------------------------------------------
template <typename AT>
__device__ __forceinline__ v16h load_frag(const AT* __restrict__ base, size_t ld,
                                          int row, int k0, int lane) {
    const AT* p = base + (size_t)row * ld + (size_t)(k0 + ((lane >> 4) << 3));
    v16h r;
#pragma unroll
    for (int i = 0; i < 8; ++i) {
        r[i]     = (_Float16)p[i];
        r[i + 8] = (_Float16)p[16 + i];
    }
    return r;
}

// ---------------------------------------------------------------------------
// TDM: stage a 16-row x K-col strip of a row-major [M,K] tensor into LDS.
// D# per CDNA5 ISA 8.3/8.4: count=1, type=2(image), data_size, 2D tile.
// Must be executed by exactly one wave (TDM ignores EXEC); caller barriers.
// ---------------------------------------------------------------------------
template <typename AT>
__device__ __forceinline__ void tdm_load_strip(const AT* gsrc, int K, int M,
                                               int m0, void* lds) {
    unsigned long long ga =
        (unsigned long long)(uintptr_t)(gsrc + (size_t)m0 * K);
    unsigned lds_addr = (unsigned)(uintptr_t)lds;      // LDS byte offset
    u32x4 g0;
    g0.x = 1u;                                         // count=1, is_restore=0
    g0.y = lds_addr;                                   // [63:32]  lds_addr
    g0.z = (unsigned)(ga & 0xFFFFFFFFull);             // [95:64]  global_addr lo
    g0.w = (unsigned)((ga >> 32) & 0x1FFFFFFull)       // [120:96] global_addr hi
         | (2u << 30);                                 // [127:126] type=2
    const unsigned dsz = (sizeof(AT) == 2) ? 1u : 2u;  // 0=1B,1=2B,2=4B
    i32x8 g1;
    g1[0] = (int)(dsz << 16);                          // wg_mask=0, data_size
    g1[1] = (int)(((unsigned)K & 0xFFFFu) << 16);      // tensor_dim0 lo16 @bit48
    g1[2] = (int)((((unsigned)K >> 16) & 0xFFFFu)      // tensor_dim0 hi16
                | (((unsigned)M & 0xFFFFu) << 16));    // tensor_dim1 lo16
    g1[3] = (int)((((unsigned)M >> 16) & 0xFFFFu)      // tensor_dim1 hi16
                | (((unsigned)K & 0xFFFFu) << 16));    // tile_dim0 = K
    g1[4] = 16;                                        // tile_dim1 = 16 rows
    g1[5] = K;                                         // tensor_dim0_stride lo
    g1[6] = 0;
    g1[7] = 0;
    i32x4 z4 = {0, 0, 0, 0};                           // groups 2/3 unused (2D)
#if defined(__clang_major__) && __clang_major__ >= 23
    i32x8 z8 = {};
    __builtin_amdgcn_tensor_load_to_lds(g0, g1, z4, z4, z8, 0);
#else
    __builtin_amdgcn_tensor_load_to_lds(g0, g1, z4, z4, 0);
#endif
    __builtin_amdgcn_s_wait_tensorcnt(0);
}

// ---------------------------------------------------------------------------
// f32 -> f16 conversion (weights)
// ---------------------------------------------------------------------------
__global__ void k_cvt(const float* __restrict__ s, _Float16* __restrict__ d, long n) {
    long i = (long)blockIdx.x * blockDim.x + threadIdx.x;
    if (i < n) d[i] = (_Float16)s[i];
}

// ---------------------------------------------------------------------------
// RMSNorm, one wave per token (256 dims -> 8 per lane).
// ---------------------------------------------------------------------------
__global__ void k_rmsnorm(const float* __restrict__ x, const float* __restrict__ w,
                          const float* __restrict__ pos,
                          _Float16* __restrict__ xn_h, _Float16* __restrict__ qk_h,
                          int ntok) {
    int wave = (int)((blockIdx.x * blockDim.x + threadIdx.x) >> 5);
    int lane = threadIdx.x & 31;
    if (wave >= ntok) return;
    const float* xr = x + (size_t)wave * D;
    float v[8];
    float ss = 0.f;
#pragma unroll
    for (int i = 0; i < 8; ++i) { v[i] = xr[lane * 8 + i]; ss += v[i] * v[i]; }
#pragma unroll
    for (int m = 16; m >= 1; m >>= 1) ss += __shfl_xor(ss, m, 32);
    float r = rsqrtf(ss * (1.0f / D) + 1e-6f);
#pragma unroll
    for (int i = 0; i < 8; ++i) {
        int c = lane * 8 + i;
        float xv = v[i] * r * w[c];
        if (xn_h) xn_h[(size_t)wave * D + c] = (_Float16)xv;
        if (qk_h) qk_h[(size_t)wave * D + c] = (_Float16)(xv + pos[(size_t)wave * D + c]);
    }
}

// ---------------------------------------------------------------------------
// WMMA GEMM, block-per-M-tile: TDM stages the 16xK A strip into LDS once,
// 8 waves cover N = TPW*128 output columns (TPW tiles each) from LDS.
// C[M,N] = A[M,K] @ W[N,K]^T + bias; optional fused dst = rbase + ls[n]*C.
// M mult of 16, K mult of 32. Grid = M/16 blocks of 256 threads.
// ---------------------------------------------------------------------------
template <typename AT, int TPW>
__global__ __launch_bounds__(256) void k_gemm(const AT* __restrict__ A,
                       const _Float16* __restrict__ W,
                       const float* __restrict__ bias,
                       float* __restrict__ Cf, _Float16* __restrict__ Ch,
                       const float* __restrict__ rbase, const float* __restrict__ ls,
                       int M, int K) {
    constexpr int N = TPW * 128;
    __shared__ __align__(16) char smem_raw[16 * 1024 * 2];   // 32 KB: 16 x K elems
    AT* As = (AT*)smem_raw;
    const int lane = threadIdx.x & 31;
    const int wave = threadIdx.x >> 5;
    const int mt = blockIdx.x;
    if (wave == 0) tdm_load_strip(A, K, M, mt * 16, As);
    __syncthreads();
    const int m0 = mt * 16 + ((lane >> 4) << 3);
#pragma unroll
    for (int j = 0; j < TPW; ++j) {
        const int nt = j * 8 + wave;
        const int bn = nt * 16 + (lane & 15);
        v8f acc = {};
        for (int k0 = 0; k0 < K; k0 += 32) {
            v16h a = load_frag(As, (size_t)K, (lane & 15), k0, lane);
            v16h b = load_frag(W, (size_t)K, bn, k0, lane);
            acc = __builtin_amdgcn_wmma_f32_16x16x32_f16(false, a, false, b,
                                                         (short)0, acc, false, false);
        }
        const int n = bn;
        float bv = bias ? bias[n] : 0.f;
#pragma unroll
        for (int r = 0; r < 8; ++r) {
            size_t off = (size_t)(m0 + r) * N + n;
            float val = acc[r] + bv;
            if (rbase) val = rbase[off] + ls[n] * val;
            if (Cf) Cf[off] = val;
            if (Ch) Ch[off] = (_Float16)val;
        }
    }
}

// ---------------------------------------------------------------------------
// Fused gated-FFN GEMM (same blocking): h = silu(A@Wg^T+bg)*(A@Wu^T+bu), f16.
// N = 1024 fixed -> 8 N-tiles per wave.
// ---------------------------------------------------------------------------
__global__ __launch_bounds__(256) void k_gemm_gate(const _Float16* __restrict__ A,
                            const _Float16* __restrict__ Wg, const _Float16* __restrict__ Wu,
                            const float* __restrict__ bg, const float* __restrict__ bu,
                            _Float16* __restrict__ Hh, int M, int K) {
    constexpr int N = DFFN;
    __shared__ __align__(16) char smem_raw[16 * 256 * 2];    // 16 x K(=256) f16
    _Float16* As = (_Float16*)smem_raw;
    const int lane = threadIdx.x & 31;
    const int wave = threadIdx.x >> 5;
    const int mt = blockIdx.x;
    if (wave == 0) tdm_load_strip(A, K, M, mt * 16, As);
    __syncthreads();
    const int m0 = mt * 16 + ((lane >> 4) << 3);
#pragma unroll
    for (int j = 0; j < 8; ++j) {
        const int nt = j * 8 + wave;
        const int bn = nt * 16 + (lane & 15);
        v8f ag = {}, au = {};
        for (int k0 = 0; k0 < K; k0 += 32) {
            v16h a  = load_frag(As, (size_t)K, (lane & 15), k0, lane);
            v16h wg = load_frag(Wg, (size_t)K, bn, k0, lane);
            v16h wu = load_frag(Wu, (size_t)K, bn, k0, lane);
            ag = __builtin_amdgcn_wmma_f32_16x16x32_f16(false, a, false, wg, (short)0, ag, false, false);
            au = __builtin_amdgcn_wmma_f32_16x16x32_f16(false, a, false, wu, (short)0, au, false, false);
        }
        float bgv = bg[bn], buv = bu[bn];
#pragma unroll
        for (int r = 0; r < 8; ++r) {
            float g = ag[r] + bgv;
            float u = au[r] + buv;
            float h = (g / (1.0f + __expf(-g))) * u;         // silu(g) * u
            Hh[(size_t)(m0 + r) * N + bn] = (_Float16)h;
        }
    }
}

// ---------------------------------------------------------------------------
// Self-attention: one wave per (b, h, 16-query block).
// Scores via WMMA (HD=32 == one K step), 900-wide strip in LDS, row softmax,
// then PV with lane == output dim (HD == wave32).
// ---------------------------------------------------------------------------
__global__ __launch_bounds__(32) void k_attention(const _Float16* __restrict__ q_h,
                                                  const _Float16* __restrict__ k_h,
                                                  const _Float16* __restrict__ v_h,
                                                  _Float16* __restrict__ sa_h) {
    __shared__ float smem[16 * 912];                 // 58.4 KB
    const int lane = threadIdx.x;
    const int qb = blockIdx.x % QBLK;
    const int bh = blockIdx.x / QBLK;
    const int h  = bh % NH;
    const int b  = bh / NH;
    const size_t ld = (size_t)BS * D;
    const _Float16* qbase = q_h + (size_t)b * D + h * HD;
    const _Float16* kbase = k_h + (size_t)b * D + h * HD;

    int qrow = qb * 16 + (lane & 15);
    if (qrow >= NQ) qrow = NQ - 1;
    v16h afrag = load_frag(qbase, ld, qrow, 0, lane);

    const float scale = 0.17677669529663689f;        // 1/sqrt(32)
    for (int kb = 0; kb < QBLK; ++kb) {
        int krow = kb * 16 + (lane & 15);
        int kc = krow < NQ ? krow : NQ - 1;
        v16h bfrag = load_frag(kbase, ld, kc, 0, lane);
        v8f c = {};
        c = __builtin_amdgcn_wmma_f32_16x16x32_f16(false, afrag, false, bfrag,
                                                   (short)0, c, false, false);
        int n = kb * 16 + (lane & 15);
        int m0 = (lane >> 4) << 3;
        bool kvalid = (n < NQ);
#pragma unroll
        for (int r = 0; r < 8; ++r)
            smem[(m0 + r) * 912 + n] = kvalid ? c[r] * scale : -INFINITY;
    }
    if (lane < 16) {                                 // intra-wave DS is in order
        float* row = smem + lane * 912;
        float mx = -INFINITY;
        for (int k = 0; k < NQ; ++k) mx = fmaxf(mx, row[k]);
        float s = 0.f;
        for (int k = 0; k < NQ; ++k) { float e = __expf(row[k] - mx); row[k] = e; s += e; }
        float inv = 1.0f / s;
        for (int k = 0; k < NQ; ++k) row[k] *= inv;
    }
    float acc[16];
#pragma unroll
    for (int i = 0; i < 16; ++i) acc[i] = 0.f;
    const int d = lane;                              // HD == wave32
    for (int k = 0; k < NQ; ++k) {
        float vv = (float)v_h[((size_t)k * BS + b) * D + h * HD + d];
#pragma unroll
        for (int qi = 0; qi < 16; ++qi) acc[qi] += smem[qi * 912 + k] * vv;
    }
#pragma unroll
    for (int qi = 0; qi < 16; ++qi) {
        int q = qb * 16 + qi;
        if (q < NQ)
            sa_h[((size_t)q * BS + b) * D + h * HD + d] = (_Float16)acc[qi];
    }
}

// ---------------------------------------------------------------------------
// Softmax over L*P=16 sampling logits per (token, head), in place.
// ---------------------------------------------------------------------------
__global__ void k_attw(float* __restrict__ logits, int n) {
    int i = blockIdx.x * blockDim.x + threadIdx.x;
    if (i >= n) return;
    float* p = logits + (size_t)i * 16;
    float mx = -INFINITY;
#pragma unroll
    for (int j = 0; j < 16; ++j) mx = fmaxf(mx, p[j]);
    float s = 0.f;
#pragma unroll
    for (int j = 0; j < 16; ++j) { float e = __expf(p[j] - mx); p[j] = e; s += e; }
    float inv = 1.0f / s;
#pragma unroll
    for (int j = 0; j < 16; ++j) p[j] *= inv;
}

// ---------------------------------------------------------------------------
// Deformable bilinear sampling: one thread per (token, head, dim).
// ---------------------------------------------------------------------------
__global__ void k_sample(const _Float16* __restrict__ val_h,
                         const float* __restrict__ off, const float* __restrict__ attw,
                         const float* __restrict__ refpts,
                         const int* __restrict__ shapes, const int* __restrict__ lstart,
                         _Float16* __restrict__ out_h) {
    int idx = blockIdx.x * blockDim.x + threadIdx.x;
    if (idx >= NTOK * D) return;
    int d = idx & 31;
    int h = (idx >> 5) & 7;
    int t = idx >> 8;              // t = q*BS + b
    int b = t & (BS - 1);
    float acc = 0.f;
    for (int l = 0; l < LL; ++l) {
        int Hl = shapes[l * 2 + 0], Wl = shapes[l * 2 + 1];
        int st = lstart[l];
        float rx = refpts[((size_t)t * LL + l) * 2 + 0];
        float ry = refpts[((size_t)t * LL + l) * 2 + 1];
        for (int p = 0; p < PP; ++p) {
            float ox = off[(size_t)t * D + h * 32 + l * 8 + p * 2 + 0];
            float oy = off[(size_t)t * D + h * 32 + l * 8 + p * 2 + 1];
            float x = (rx + ox / (float)Wl) * Wl - 0.5f;
            float y = (ry + oy / (float)Hl) * Hl - 0.5f;
            float x0 = floorf(x), y0 = floorf(y);
            float s = 0.f;
#pragma unroll
            for (int dx = 0; dx < 2; ++dx)
#pragma unroll
                for (int dy = 0; dy < 2; ++dy) {
                    float xi = x0 + dx, yi = y0 + dy;
                    if (xi >= 0.f && xi <= (float)(Wl - 1) && yi >= 0.f && yi <= (float)(Hl - 1)) {
                        float w = (1.0f - fabsf(x - xi)) * (1.0f - fabsf(y - yi));
                        size_t vi = ((size_t)(st + (int)yi * Wl + (int)xi) * BS + b) * D + h * HD + d;
                        s += w * (float)val_h[vi];
                    }
                }
            acc += attw[(size_t)t * 128 + h * 16 + l * 4 + p] * s;
        }
    }
    out_h[(size_t)t * D + h * HD + d] = (_Float16)acc;
}

// ---------------------------------------------------------------------------
// Host orchestration
// ---------------------------------------------------------------------------
static inline char* bump(char*& p, size_t bytes) {
    char* r = p;
    p += (bytes + 255) & ~(size_t)255;
    return r;
}

extern "C" void kernel_launch(void* const* d_in, const int* in_sizes, int n_in,
                              void* d_out, int out_size, void* d_ws, size_t ws_size,
                              hipStream_t stream) {
    (void)in_sizes; (void)n_in; (void)out_size; (void)ws_size;
    const float* tgt   = (const float*)d_in[0];
    const float* pos   = (const float*)d_in[1];
    const float* refp  = (const float*)d_in[2];
    const float* mem   = (const float*)d_in[3];
    const int*   shp   = (const int*)d_in[4];
    const int*   lsi   = (const int*)d_in[5];
    const float* w_sa  = (const float*)d_in[6];
    const float* w_ca  = (const float*)d_in[7];
    const float* w_ffn = (const float*)d_in[8];
    const float* ls_sa = (const float*)d_in[9];
    const float* ls_ca = (const float*)d_in[10];
    const float* ls_ff = (const float*)d_in[11];
    const float* Wq = (const float*)d_in[12];
    const float* Wk = (const float*)d_in[13];
    const float* Wv = (const float*)d_in[14];
    const float* bq = (const float*)d_in[15];
    const float* bk = (const float*)d_in[16];
    const float* bv = (const float*)d_in[17];
    const float* Wo_sa = (const float*)d_in[18];
    const float* bo_sa = (const float*)d_in[19];
    const float* Wvp = (const float*)d_in[20];
    const float* bvp = (const float*)d_in[21];
    const float* Woff = (const float*)d_in[22];
    const float* boff = (const float*)d_in[23];
    const float* Wat = (const float*)d_in[24];
    const float* bat = (const float*)d_in[25];
    const float* Wout = (const float*)d_in[26];
    const float* bout = (const float*)d_in[27];
    const float* Wg = (const float*)d_in[28];
    const float* bg = (const float*)d_in[29];
    const float* Wu = (const float*)d_in[30];
    const float* bu = (const float*)d_in[31];
    const float* Wf = (const float*)d_in[32];
    const float* bf = (const float*)d_in[33];
    float* out = (float*)d_out;

    // ---- workspace layout ----
    char* p = (char*)d_ws;
    _Float16* wq_h  = (_Float16*)bump(p, (size_t)D * D * 2);
    _Float16* wk_h  = (_Float16*)bump(p, (size_t)D * D * 2);
    _Float16* wv_h  = (_Float16*)bump(p, (size_t)D * D * 2);
    _Float16* wo_h  = (_Float16*)bump(p, (size_t)D * D * 2);
    _Float16* wvp_h = (_Float16*)bump(p, (size_t)D * D * 2);
    _Float16* wof_h = (_Float16*)bump(p, (size_t)D * D * 2);
    _Float16* wat_h = (_Float16*)bump(p, (size_t)128 * D * 2);
    _Float16* wou_h = (_Float16*)bump(p, (size_t)D * D * 2);
    _Float16* wg_h  = (_Float16*)bump(p, (size_t)DFFN * D * 2);
    _Float16* wu_h  = (_Float16*)bump(p, (size_t)DFFN * D * 2);
    _Float16* wf_h  = (_Float16*)bump(p, (size_t)D * DFFN * 2);
    _Float16* xn_h  = (_Float16*)bump(p, (size_t)NTOK * D * 2);
    _Float16* qk_h  = (_Float16*)bump(p, (size_t)NTOK * D * 2);   // qk / qca
    _Float16* q_h   = (_Float16*)bump(p, (size_t)NTOK * D * 2);
    _Float16* k_hh  = (_Float16*)bump(p, (size_t)NTOK * D * 2);
    _Float16* v_hh  = (_Float16*)bump(p, (size_t)NTOK * D * 2);
    _Float16* sa_h  = (_Float16*)bump(p, (size_t)NTOK * D * 2);   // SA out / CA samples
    _Float16* val_h = (_Float16*)bump(p, (size_t)NVTOK * D * 2);
    _Float16* hh    = (_Float16*)bump(p, (size_t)NTOK * DFFN * 2);
    float*    off_f = (float*)bump(p, (size_t)NTOK * D * 4);
    float*    aw_f  = (float*)bump(p, (size_t)NTOK * 128 * 4);

    // ---- weight conversion ----
    struct { const float* s; _Float16* d; long n; } cv[] = {
        {Wq, wq_h, D * D}, {Wk, wk_h, D * D}, {Wv, wv_h, D * D}, {Wo_sa, wo_h, D * D},
        {Wvp, wvp_h, D * D}, {Woff, wof_h, D * D}, {Wat, wat_h, 128 * D},
        {Wout, wou_h, D * D}, {Wg, wg_h, (long)DFFN * D}, {Wu, wu_h, (long)DFFN * D},
        {Wf, wf_h, (long)D * DFFN}};
    for (auto& c : cv)
        k_cvt<<<(int)((c.n + 255) / 256), 256, 0, stream>>>(c.s, c.d, c.n);

    // ---- self-attention block ----
    k_rmsnorm<<<NTOK / 8, 256, 0, stream>>>(tgt, w_sa, pos, xn_h, qk_h, NTOK);
    k_gemm<_Float16, 2><<<NTOK / 16, 256, 0, stream>>>(
        qk_h, wq_h, bq, nullptr, q_h, nullptr, nullptr, NTOK, D);
    k_gemm<_Float16, 2><<<NTOK / 16, 256, 0, stream>>>(
        qk_h, wk_h, bk, nullptr, k_hh, nullptr, nullptr, NTOK, D);
    k_gemm<_Float16, 2><<<NTOK / 16, 256, 0, stream>>>(
        xn_h, wv_h, bv, nullptr, v_hh, nullptr, nullptr, NTOK, D);
    k_attention<<<QBLK * BS * NH, 32, 0, stream>>>(q_h, k_hh, v_hh, sa_h);
    k_gemm<_Float16, 2><<<NTOK / 16, 256, 0, stream>>>(
        sa_h, wo_h, bo_sa, out, nullptr, tgt, ls_sa, NTOK, D);      // tgt1 -> d_out

    // ---- deformable cross-attention block ----
    k_rmsnorm<<<NTOK / 8, 256, 0, stream>>>(out, w_ca, pos, nullptr, qk_h, NTOK);
    k_gemm<float, 2><<<NVTOK / 16, 256, 0, stream>>>(
        mem, wvp_h, bvp, nullptr, val_h, nullptr, nullptr, NVTOK, D);
    k_gemm<_Float16, 2><<<NTOK / 16, 256, 0, stream>>>(
        qk_h, wof_h, boff, off_f, nullptr, nullptr, nullptr, NTOK, D);
    k_gemm<_Float16, 1><<<NTOK / 16, 256, 0, stream>>>(
        qk_h, wat_h, bat, aw_f, nullptr, nullptr, nullptr, NTOK, D);
    k_attw<<<(NTOK * NH + 255) / 256, 256, 0, stream>>>(aw_f, NTOK * NH);
    k_sample<<<(NTOK * D + 255) / 256, 256, 0, stream>>>(
        val_h, off_f, aw_f, refp, shp, lsi, sa_h);
    k_gemm<_Float16, 2><<<NTOK / 16, 256, 0, stream>>>(
        sa_h, wou_h, bout, out, nullptr, out, ls_ca, NTOK, D);      // tgt2 -> d_out

    // ---- gated FFN block ----
    k_rmsnorm<<<NTOK / 8, 256, 0, stream>>>(out, w_ffn, nullptr, xn_h, nullptr, NTOK);
    k_gemm_gate<<<NTOK / 16, 256, 0, stream>>>(xn_h, wg_h, wu_h, bg, bu, hh, NTOK, D);
    k_gemm<_Float16, 2><<<NTOK / 16, 256, 0, stream>>>(
        hh, wf_h, bf, out, nullptr, out, ls_ff, NTOK, DFFN);        // final -> d_out
}
